// ITPNet_35338990911516
// MI455X (gfx1250) — matrix operand order repaired and checked by reading, not compiled, not run on hardware
//
#include <hip/hip_runtime.h>
#include <math.h>

namespace {

constexpr int kF = 128;      // feature dim
constexpr int kK = 32;       // radial basis count
constexpr int kN = 25000;    // atoms
constexpr int kE = 200000;   // edges
constexpr int kG = 64;       // graphs
constexpr float INV_SQRT2 = 0.70710678118654752f;
constexpr float INV_SQRT3 = 0.57735026918962576f;
constexpr float INV_SQRTF = 0.08838834764831845f;  // 1/sqrt(128)

constexpr int LSA = 136;     // LDS row stride (halfs) for 128-half rows: banks 4*ln, conflict-free
constexpr int LSE = 40;      // LDS row stride (halfs) for 32-half rows: banks 20*ln mod 64, conflict-free

typedef __attribute__((ext_vector_type(16))) _Float16 v16h;
typedef __attribute__((ext_vector_type(8)))  _Float16 v8h;
typedef __attribute__((ext_vector_type(8)))  float    v8f;
typedef __attribute__((__vector_size__(16))) int      v4i_;
typedef _Float16 h16;

#if defined(__has_builtin)
#if __has_builtin(__builtin_amdgcn_global_load_async_to_lds_b128) && \
    __has_builtin(__builtin_amdgcn_s_wait_asynccnt)
#define ITP_ASYNC_LDS 1
#endif
#endif
#ifndef ITP_ASYNC_LDS
#define ITP_ASYNC_LDS 0
#endif

// ---------------------------------------------------------------------------
// helpers
// ---------------------------------------------------------------------------
__device__ __forceinline__ void atomAddF(float* p, float v) {
  __hip_atomic_fetch_add(p, v, __ATOMIC_RELAXED, __HIP_MEMORY_SCOPE_AGENT);
}

__device__ __forceinline__ float fastRcp(float x)  { return __builtin_amdgcn_rcpf(x); }
__device__ __forceinline__ float fastRsq(float x)  { return __builtin_amdgcn_rsqf(x); }
__device__ __forceinline__ float fastSigmoid(float x) {
  return fastRcp(1.f + __expf(-x));      // v_exp + v_add + v_rcp
}

__device__ __forceinline__ float blockReduce128(float v, float* sh, int tid) {
  sh[tid] = v;
  __syncthreads();
#pragma unroll
  for (int s = 64; s > 0; s >>= 1) {
    if (tid < s) sh[tid] += sh[tid + s];
    __syncthreads();
  }
  float r = sh[0];
  __syncthreads();
  return r;
}

// Stage a ROWS x GS (halfs) tile from global into LDS with padded row stride LS.
// Uses the CDNA5 async global->LDS data path when available (ASYNCcnt +
// s_wait_asynccnt), else a VGPR bounce. Followed by a workgroup barrier.
template <int ROWS, int GS, int LS>
__device__ __forceinline__ void stage_tile(const h16* __restrict__ g, h16* l, int tid) {
  constexpr int CPR = GS / 8;          // 16-byte chunks per row
  constexpr int TOT = ROWS * CPR;
  if (tid < TOT) {
    const int r = tid / CPR, c = tid % CPR;
    const h16* gp = g + (size_t)r * GS + c * 8;
    h16* lp = l + r * LS + c * 8;
#if ITP_ASYNC_LDS
    // arg0: generic v4i* (matches this toolchain's prototype);
    // arg1: AS(3)-qualified v4i* (exact if param is AS3; implicitly converts
    //       to generic otherwise, since generic is a superset of LDS).
    __builtin_amdgcn_global_load_async_to_lds_b128(
        (v4i_*)gp,
        (__attribute__((address_space(3))) v4i_*)lp, 0, 0);
#else
    *(v8h*)lp = *(const v8h*)gp;
#endif
  }
#if ITP_ASYNC_LDS
  __builtin_amdgcn_s_wait_asynccnt(0);
#endif
  __syncthreads();
}

// A-operand fragment for v_wmma_f32_16x16x32_f16 read from staged LDS tile.
// Lane L: row m = L&15, half h = L>>4. a[0..15] = K(16h..16h+15)
__device__ __forceinline__ v16h loadA_lds(const h16* sA, int ln, int kbase, int h, int lstride) {
  const v8h lo = *(const v8h*)(sA + ln * lstride + kbase + h * 16);
  const v8h hi = *(const v8h*)(sA + ln * lstride + kbase + h * 16 + 8);
  v16h a;
#pragma unroll
  for (int j = 0; j < 8; ++j) { a[j] = lo[j]; a[j + 8] = hi[j]; }
  return a;
}

// ---------------------------------------------------------------------------
// weight prep: f32 -> transposed f16 (Wt[n*K + k] = W[k*F + n])
// ---------------------------------------------------------------------------
struct WSet { const float* m[20]; };

__global__ void prep_weights(WSet ws, h16* __restrict__ wt, h16* __restrict__ wbt) {
  const int b = blockIdx.x;
  if (b < 18) {
    const float* W = ws.m[b];
    h16* dst = wt + (size_t)b * kF * kF;
    for (int i = threadIdx.x; i < kF * kF; i += blockDim.x) {
      const int n = i >> 7, k = i & 127;
      dst[i] = (h16)W[k * kF + n];
    }
  } else {
    const float* W = ws.m[b];                        // 32 x 128
    h16* dst = wbt + (size_t)(b - 18) * kF * kK;
    for (int i = threadIdx.x; i < kF * kK; i += blockDim.x) {
      const int n = i / kK, k = i % kK;
      dst[i] = (h16)W[k * kF + n];
    }
  }
}

// ---------------------------------------------------------------------------
// generic WMMA GEMM: C[M,128] = A[M,128](f16) @ Wt(128x128, transposed f16) + bias
// block = 256 threads = 8 waves; A tile staged once in LDS (async path),
// each wave owns one 16-col N tile. EPI: 0 = f32 out, 1 = f16 out, 2 = both
// ---------------------------------------------------------------------------
template <int EPI>
__global__ void gemm128(const h16* __restrict__ A, const h16* __restrict__ Wt,
                        const float* __restrict__ bias,
                        float* __restrict__ Cf, h16* __restrict__ Ch, int M) {
  __shared__ __align__(16) h16 sA[16 * LSA];
  const int tid  = threadIdx.x;
  const int lane = tid & 31;
  const int wv   = tid >> 5;
  const int h    = lane >> 4;
  const int ln   = lane & 15;
  const int n    = wv * 16 + ln;

  stage_tile<16, kF, LSA>(A + (size_t)blockIdx.x * 16 * kF, sA, tid);

  v8f acc = {};
#pragma unroll
  for (int kk = 0; kk < 4; ++kk) {
    v16h a = loadA_lds(sA, ln, kk * 32, h, LSA);
    // B lane: col n, b[j] = Wt[n][kk*32 + 16h + j]  (one 32B load)
    v16h b = *(const v16h*)(Wt + (size_t)n * kF + kk * 32 + h * 16);
    acc = __builtin_amdgcn_wmma_f32_16x16x32_f16(false, a, false, b, (short)0, acc,
                                                 false, false);
  }
  const float bn = bias ? bias[n] : 0.f;
#pragma unroll
  for (int i = 0; i < 8; ++i) {
    const int row = blockIdx.x * 16 + h * 8 + i;
    if (row < M) {
      const float v = acc[i] + bn;
      if (EPI == 0 || EPI == 2) Cf[(size_t)row * kF + n] = v;
      if (EPI == 1 || EPI == 2) Ch[(size_t)row * kF + n] = (h16)v;
    }
  }
}

// ---------------------------------------------------------------------------
// embedding gather: ez16 = (f16) embed_Z[z], s0 = embed_Z[z]
// ---------------------------------------------------------------------------
__global__ void gather_embed(const float* __restrict__ emb, const int* __restrict__ z,
                             h16* __restrict__ ez16, float* __restrict__ s0) {
  const int idx = blockIdx.x * blockDim.x + threadIdx.x;
  if (idx >= kN * kF) return;
  const int a = idx >> 7, f = idx & 127;
  const float v = emb[(size_t)z[a] * kF + f];
  ez16[idx] = (h16)v;
  s0[idx]   = v;
}

// ---------------------------------------------------------------------------
// deloc_embed pieces (segment softmax over G graphs)
// ---------------------------------------------------------------------------
__global__ void init_gstats(unsigned* __restrict__ gmaxU, float* __restrict__ gsum) {
  const int i = threadIdx.x;
  if (i < kG) { gmaxU[i] = 0u; gsum[i] = 0.f; }  // key 0 < enc(any float)
}

__global__ void deloc_logits(const float* __restrict__ q, const float* __restrict__ kp,
                             const float* __restrict__ km, const float* __restrict__ psi,
                             const int* __restrict__ segs, float* __restrict__ logits,
                             unsigned* __restrict__ gmaxU) {
  __shared__ float sh[kF];
  const int a = blockIdx.x, f = threadIdx.x;
  const int sg = segs[a];
  const float pv = psi[sg];
  const float kkv = (pv >= 0.f) ? kp[f] : km[f];
  const float s = blockReduce128(q[(size_t)a * kF + f] * kkv, sh, f);
  if (f == 0) {
    const float lg = s * INV_SQRTF;
    logits[a] = lg;
    const unsigned u = __float_as_uint(lg);
    const unsigned key = (u & 0x80000000u) ? ~u : (u | 0x80000000u);
    atomicMax(gmaxU + sg, key);
  }
}

__global__ void deloc_ex(float* __restrict__ logits, const unsigned* __restrict__ gmaxU,
                         float* __restrict__ gsum, const int* __restrict__ segs) {
  const int a = blockIdx.x * blockDim.x + threadIdx.x;
  if (a >= kN) return;
  const int sg = segs[a];
  const unsigned key = gmaxU[sg];
  const float m = (key & 0x80000000u) ? __uint_as_float(key & 0x7FFFFFFFu)
                                      : __uint_as_float(~key);
  const float ex = __expf(logits[a] - m);
  logits[a] = ex;
  atomAddF(gsum + sg, ex);
}

template <bool WRITE_H>
__global__ void deloc_accum(const float* __restrict__ ex, const float* __restrict__ gsum,
                            const int* __restrict__ segs, const float* __restrict__ psi,
                            const float* __restrict__ vbuf, float* __restrict__ s0,
                            h16* __restrict__ s016) {
  const int a = blockIdx.x, f = threadIdx.x;
  const int sg = segs[a];
  const float amt = psi[sg] * ex[a] * fastRcp(gsum[sg] + 1e-12f);
  const size_t idx = (size_t)a * kF + f;
  const float v = s0[idx] + amt * vbuf[idx];
  s0[idx] = v;
  if (WRITE_H) s016[idx] = (h16)v;
}

// ---------------------------------------------------------------------------
// edge geometry: unit vectors + Bernstein radial basis * smooth cutoff (f16)
// ---------------------------------------------------------------------------
__global__ void edge_geom(const float* __restrict__ pos, const int* __restrict__ src,
                          const int* __restrict__ dst, h16* __restrict__ w16,
                          float* __restrict__ uvec) {
  const int e = blockIdx.x * blockDim.x + threadIdx.x;
  if (e >= kE) return;
  const int s = src[e], d = dst[e];
  const float dx = pos[s * 3 + 0] - pos[d * 3 + 0];
  const float dy = pos[s * 3 + 1] - pos[d * 3 + 1];
  const float dz = pos[s * 3 + 2] - pos[d * 3 + 2];
  const float r = sqrtf(dx * dx + dy * dy + dz * dz + 1e-12f);
  const float inv = fastRcp(r);
  uvec[e * 3 + 0] = dx * inv;
  uvec[e * 3 + 1] = dy * inv;
  uvec[e * 3 + 2] = dz * inv;
  float t = __expf(-0.5f * r);
  t = fminf(fmaxf(t, 1e-6f), 1.f - 1e-6f);
  const float rc = r * 0.2f;
  const float x2 = fminf(rc * rc, 1.f - 1e-6f);
  const float cut = (r < 5.f) ? __expf(1.f - fastRcp(1.f - x2)) : 0.f;
  const float omt = 1.f - t;
  float p = __powf(omt, 31.f);       // (1-t)^(K-1)
  const float ratio = t * fastRcp(omt);
  float binc = 1.f;                  // C(31, k)
#pragma unroll
  for (int k = 0; k < kK; ++k) {
    w16[(size_t)e * kK + k] = (h16)(binc * p * cut);
    p *= ratio;
    binc *= (float)(31 - k) / (float)(k + 1);
  }
}

// ---------------------------------------------------------------------------
// fused edge block: basis GEMMs (K=32, WMMA) + bias + SiLU gate + tensor
// product with gathered s0[src] + scatter-add into y (f32 atomics).
// grid.x = E/16 edge tiles; 8 waves cover 128 feature columns; the 16x32
// basis tile is staged once in LDS via the async path.
// ---------------------------------------------------------------------------
__global__ void edge_msg(const h16* __restrict__ w16, const h16* __restrict__ wbt0,
                         const h16* __restrict__ wbt1, const float* __restrict__ bb0,
                         const float* __restrict__ uvec, const int* __restrict__ src,
                         const int* __restrict__ dst, const float* __restrict__ s0,
                         float* __restrict__ ys, float* __restrict__ yv0,
                         float* __restrict__ yv1, float* __restrict__ yv2) {
  __shared__ __align__(16) h16 sA[16 * LSE];
  const int tid  = threadIdx.x;
  const int lane = tid & 31;
  const int wv   = tid >> 5;
  const int h    = lane >> 4;
  const int ln   = lane & 15;
  const int n    = wv * 16 + ln;
  const int e0   = blockIdx.x * 16;

  stage_tile<16, kK, LSE>(w16 + (size_t)e0 * kK, sA, tid);

  const v16h a  = loadA_lds(sA, ln, 0, h, LSE);
  const v16h b0 = *(const v16h*)(wbt0 + (size_t)n * kK + h * 16);
  const v16h b1 = *(const v16h*)(wbt1 + (size_t)n * kK + h * 16);
  v8f c0 = {}, c1 = {};
  c0 = __builtin_amdgcn_wmma_f32_16x16x32_f16(false, a, false, b0, (short)0, c0, false, false);
  c1 = __builtin_amdgcn_wmma_f32_16x16x32_f16(false, a, false, b1, (short)0, c1, false, false);

  const float bn = bb0[n];
#pragma unroll
  for (int i = 0; i < 8; ++i) {
    const int e = e0 + h * 8 + i;
    const float P = c0[i] + bn;
    const float g = fastSigmoid(P);                // SiLU gate
    const float Pg = P * g;
    const float Qg = c1[i] * g;
    const int s = src[e], d = dst[e];
    const float sv = s0[(size_t)s * kF + n];
    atomAddF(ys + (size_t)d * kF + n, sv * Pg);
    const float q = sv * Qg * INV_SQRT2;
    atomAddF(yv0 + (size_t)d * kF + n, q * uvec[e * 3 + 0]);
    atomAddF(yv1 + (size_t)d * kF + n, q * uvec[e * 3 + 1]);
    atomAddF(yv2 + (size_t)d * kF + n, q * uvec[e * 3 + 2]);
  }
}

// ---------------------------------------------------------------------------
// layernorms / tensor products
// ---------------------------------------------------------------------------
__global__ void ln1_kernel(const float* __restrict__ ys, const float* __restrict__ yv0,
                           const float* __restrict__ yv1, const float* __restrict__ yv2,
                           const float* __restrict__ g0, const float* __restrict__ b0,
                           const float* __restrict__ g1, h16* __restrict__ os,
                           h16* __restrict__ ov0, h16* __restrict__ ov1,
                           h16* __restrict__ ov2) {
  __shared__ float sh[kF];
  const int a = blockIdx.x, f = threadIdx.x;
  const size_t idx = (size_t)a * kF + f;
  const float s = ys[idx];
  const float v0 = yv0[idx], v1 = yv1[idx], v2 = yv2[idx];
  const float mu  = blockReduce128(s, sh, f) * (1.f / kF);
  const float d   = s - mu;
  const float var = blockReduce128(d * d, sh, f) * (1.f / kF);
  const float n2  = blockReduce128(v0 * v0 + v1 * v1 + v2 * v2, sh, f) * (1.f / kF);
  const float isd = fastRsq(var + 1e-6f);
  const float gg  = fastRsq(n2 + 1e-6f) * g1[f];
  os[idx]  = (h16)(d * isd * g0[f] + b0[f]);
  ov0[idx] = (h16)(v0 * gg);
  ov1[idx] = (h16)(v1 * gg);
  ov2[idx] = (h16)(v2 * gg);
}

__global__ void ln2_add_kernel(const float* __restrict__ t0, const float* __restrict__ t1,
                               const float* __restrict__ t2, const float* __restrict__ t3,
                               const float* __restrict__ g0, const float* __restrict__ b0,
                               const float* __restrict__ g1, float* __restrict__ y2s,
                               const float* __restrict__ y2v0, const float* __restrict__ y2v1,
                               const float* __restrict__ y2v2, h16* __restrict__ y3s,
                               h16* __restrict__ y3v0, h16* __restrict__ y3v1,
                               h16* __restrict__ y3v2) {
  __shared__ float sh[kF];
  const int a = blockIdx.x, f = threadIdx.x;
  const size_t idx = (size_t)a * kF + f;
  const float s = t0[idx];
  const float v0 = t1[idx], v1 = t2[idx], v2 = t3[idx];
  const float mu  = blockReduce128(s, sh, f) * (1.f / kF);
  const float d   = s - mu;
  const float var = blockReduce128(d * d, sh, f) * (1.f / kF);
  const float n2  = blockReduce128(v0 * v0 + v1 * v1 + v2 * v2, sh, f) * (1.f / kF);
  const float isd = fastRsq(var + 1e-6f);
  const float gg  = fastRsq(n2 + 1e-6f) * g1[f];
  const float ns  = y2s[idx] + (d * isd * g0[f] + b0[f]);   // y3 = y2 + LN(out)
  y2s[idx] = ns;
  y3s[idx]  = (h16)ns;
  y3v0[idx] = (h16)(y2v0[idx] + v0 * gg);
  y3v1[idx] = (h16)(y2v1[idx] + v1 * gg);
  y3v2[idx] = (h16)(y2v2[idx] + v2 * gg);
}

__global__ void tp_kernel(const h16* __restrict__ as, const h16* __restrict__ av0,
                          const h16* __restrict__ av1, const h16* __restrict__ av2,
                          const h16* __restrict__ bs, const h16* __restrict__ bv0,
                          const h16* __restrict__ bv1, const h16* __restrict__ bv2,
                          h16* __restrict__ ts, h16* __restrict__ tv0,
                          h16* __restrict__ tv1, h16* __restrict__ tv2) {
  const int idx = blockIdx.x * blockDim.x + threadIdx.x;
  if (idx >= kN * kF) return;
  const float A = (float)as[idx], B = (float)bs[idx];
  const float a0 = (float)av0[idx], a1 = (float)av1[idx], a2 = (float)av2[idx];
  const float b0 = (float)bv0[idx], b1 = (float)bv1[idx], b2 = (float)bv2[idx];
  ts[idx]  = (h16)(A * B + (a0 * b0 + a1 * b1 + a2 * b2) * INV_SQRT3);
  tv0[idx] = (h16)((A * b0 + a0 * B) * INV_SQRT2);
  tv1[idx] = (h16)((A * b1 + a1 * B) * INV_SQRT2);
  tv2[idx] = (h16)((A * b2 + a2 * B) * INV_SQRT2);
}

__global__ void ts_kernel(const h16* __restrict__ as, const h16* __restrict__ av0,
                          const h16* __restrict__ av1, const h16* __restrict__ av2,
                          const h16* __restrict__ bs, const h16* __restrict__ bv0,
                          const h16* __restrict__ bv1, const h16* __restrict__ bv2,
                          h16* __restrict__ ts) {
  const int idx = blockIdx.x * blockDim.x + threadIdx.x;
  if (idx >= kN * kF) return;
  const float A = (float)as[idx], B = (float)bs[idx];
  const float a0 = (float)av0[idx], a1 = (float)av1[idx], a2 = (float)av2[idx];
  const float b0 = (float)bv0[idx], b1 = (float)bv1[idx], b2 = (float)bv2[idx];
  ts[idx] = (h16)(A * B + (a0 * b0 + a1 * b1 + a2 * b2) * INV_SQRT3);
}

__global__ void ln3_final_kernel(const float* __restrict__ outs, const float* __restrict__ g,
                                 const float* __restrict__ b, const float* __restrict__ y3s,
                                 float* __restrict__ out) {
  __shared__ float sh[kF];
  const int a = blockIdx.x, f = threadIdx.x;
  const size_t idx = (size_t)a * kF + f;
  const float s = outs[idx];
  const float mu  = blockReduce128(s, sh, f) * (1.f / kF);
  const float d   = s - mu;
  const float var = blockReduce128(d * d, sh, f) * (1.f / kF);
  out[idx] = y3s[idx] + d * fastRsq(var + 1e-6f) * g[f] + b[f];
}

}  // namespace

// ---------------------------------------------------------------------------
// host launch
// ---------------------------------------------------------------------------
extern "C" void kernel_launch(void* const* d_in, const int* in_sizes, int n_in,
                              void* d_out, int out_size, void* d_ws, size_t ws_size,
                              hipStream_t stream) {
  (void)in_sizes; (void)n_in; (void)out_size; (void)ws_size;

  // ---- input indices (setup_inputs dict order, params flattened in order) ----
  enum {
    IN_POS = 0,
    P_EMBED_Z, P_Q_WQ, P_Q_KP, P_Q_KM, P_Q_WV,
    P_S_WQ, P_S_KP, P_S_KM, P_S_WV,
    P_WB0, P_BB0, P_WB1,
    P_WZ0, P_BZ0, P_WZ1,
    P_N1_G0, P_N1_B0, P_N1_G1,
    P_WD0, P_BD0, P_WD1,
    P_I0_A0, P_I0_AB, P_I0_A1, P_I0_B0, P_I0_BB, P_I0_B1, P_I0_C0, P_I0_CB, P_I0_C1,
    P_N2_G0, P_N2_B0, P_N2_G1,
    P_I1_A0, P_I1_AB, P_I1_A1, P_I1_B0, P_I1_BB, P_I1_B1, P_I1_C0, P_I1_CB,
    P_N3_G0, P_N3_B0,
    IN_Z, IN_DST, IN_SRC, IN_NUE, IN_TC, IN_SEGS, IN_GMASK
  };
  auto fin = [&](int i) { return (const float*)d_in[i]; };
  auto iin = [&](int i) { return (const int*)d_in[i]; };

  // ---- transposed-f16 weight slots ----
  enum { W_QWQ = 0, W_QWV, W_SWQ, W_SWV, W_WZ0, W_WD0, W_WD1,
         W_I0A0, W_I0A1, W_I0B0, W_I0B1, W_I0C0, W_I0C1,
         W_I1A0, W_I1A1, W_I1B0, W_I1B1, W_I1C0, NW };
  static const int wsrc[18] = { P_Q_WQ, P_Q_WV, P_S_WQ, P_S_WV, P_WZ0, P_WD0, P_WD1,
                                P_I0_A0, P_I0_A1, P_I0_B0, P_I0_B1, P_I0_C0, P_I0_C1,
                                P_I1_A0, P_I1_A1, P_I1_B0, P_I1_B1, P_I1_C0 };

  // ---- workspace carve-up ----
  char* wp = (char*)d_ws;
  auto alloc = [&](size_t bytes) -> void* {
    void* r = (void*)wp;
    wp += (bytes + 255) & ~(size_t)255;
    return r;
  };
  const size_t NF = (size_t)kN * kF;
  h16* wt   = (h16*)alloc((size_t)NW * kF * kF * sizeof(h16));
  h16* wbt  = (h16*)alloc((size_t)2 * kF * kK * sizeof(h16));
  h16* ez16 = (h16*)alloc(NF * sizeof(h16));
  h16* s016 = (h16*)alloc(NF * sizeof(h16));
  h16* hl[4]; for (int i = 0; i < 4; ++i) hl[i] = (h16*)alloc(NF * sizeof(h16)); // yln / tp / ts
  h16* hy[4]; for (int i = 0; i < 4; ++i) hy[i] = (h16*)alloc(NF * sizeof(h16)); // y2 / y3 f16
  h16* ha[4]; for (int i = 0; i < 4; ++i) ha[i] = (h16*)alloc(NF * sizeof(h16));
  h16* hb[4]; for (int i = 0; i < 4; ++i) hb[i] = (h16*)alloc(NF * sizeof(h16));
  float* s0 = (float*)alloc(NF * sizeof(float));
  float* t[4]; for (int i = 0; i < 4; ++i) t[i] = (float*)alloc(NF * sizeof(float));
  float* ys = (float*)alloc(NF * sizeof(float));                 // y scalar, later y2s/y3s
  float* yv[3]; for (int i = 0; i < 3; ++i) yv[i] = (float*)alloc(NF * sizeof(float));
  float*    logits = (float*)alloc(kN * sizeof(float));
  unsigned* gmaxU  = (unsigned*)alloc(kG * sizeof(unsigned));
  float*    gsum   = (float*)alloc(kG * sizeof(float));
  h16*   w16  = (h16*)alloc((size_t)kE * kK * sizeof(h16));
  float* uvec = (float*)alloc((size_t)kE * 3 * sizeof(float));

  const int MT = (kN + 15) / 16;                 // 1563 M-tiles
  auto wtp = [&](int i) { return wt + (size_t)i * kF * kF; };

  // 1) weights -> transposed f16
  WSet wset;
  for (int i = 0; i < 18; ++i) wset.m[i] = fin(wsrc[i]);
  wset.m[18] = fin(P_WB0);
  wset.m[19] = fin(P_WB1);
  prep_weights<<<20, 256, 0, stream>>>(wset, wt, wbt);

  // 2) atom embeddings
  gather_embed<<<(int)((NF + 255) / 256), 256, 0, stream>>>(fin(P_EMBED_Z), iin(IN_Z), ez16, s0);

  // 3) deloc embeddings (charge then spin): q/v GEMMs + segment softmax
  for (int pass = 0; pass < 2; ++pass) {
    const int wq = pass == 0 ? W_QWQ : W_SWQ;
    const int wv_ = pass == 0 ? W_QWV : W_SWV;
    const float* kp  = fin(pass == 0 ? P_Q_KP : P_S_KP);
    const float* km  = fin(pass == 0 ? P_Q_KM : P_S_KM);
    const float* psi = fin(pass == 0 ? IN_TC : IN_NUE);
    gemm128<0><<<MT, 256, 0, stream>>>(ez16, wtp(wq), nullptr, t[0], nullptr, kN);
    gemm128<0><<<MT, 256, 0, stream>>>(ez16, wtp(wv_), nullptr, t[1], nullptr, kN);
    init_gstats<<<1, kG, 0, stream>>>(gmaxU, gsum);
    deloc_logits<<<kN, kF, 0, stream>>>(t[0], kp, km, psi, iin(IN_SEGS), logits, gmaxU);
    deloc_ex<<<(kN + 255) / 256, 256, 0, stream>>>(logits, gmaxU, gsum, iin(IN_SEGS));
    if (pass == 0)
      deloc_accum<false><<<kN, kF, 0, stream>>>(logits, gsum, iin(IN_SEGS), psi, t[1], s0, nullptr);
    else
      deloc_accum<true><<<kN, kF, 0, stream>>>(logits, gsum, iin(IN_SEGS), psi, t[1], s0, s016);
  }

  // 4) y scalar init = s0 @ Wz0 + bz0 ; y vectors zero
  gemm128<0><<<MT, 256, 0, stream>>>(s016, wtp(W_WZ0), fin(P_BZ0), ys, nullptr, kN);
  for (int c = 0; c < 3; ++c) (void)hipMemsetAsync(yv[c], 0, NF * sizeof(float), stream);

  // 5) edge geometry + fused edge message block (WMMA + SiLU + scatter atomics)
  edge_geom<<<(kE + 255) / 256, 256, 0, stream>>>(fin(IN_POS), iin(IN_SRC), iin(IN_DST), w16, uvec);
  edge_msg<<<kE / 16, 256, 0, stream>>>(w16, wbt, wbt + (size_t)kF * kK, fin(P_BB0), uvec,
                                        iin(IN_SRC), iin(IN_DST), s0, ys, yv[0], yv[1], yv[2]);

  // 6) LN1 -> f16
  ln1_kernel<<<kN, kF, 0, stream>>>(ys, yv[0], yv[1], yv[2], fin(P_N1_G0), fin(P_N1_B0),
                                    fin(P_N1_G1), hl[0], hl[1], hl[2], hl[3]);

  // 7) y2 = eqv_dense(LN1, Wd)  (f32 residual in ys/yv, f16 copies in hy)
  gemm128<2><<<MT, 256, 0, stream>>>(hl[0], wtp(W_WD0), fin(P_BD0), ys, hy[0], kN);
  for (int c = 0; c < 3; ++c)
    gemm128<2><<<MT, 256, 0, stream>>>(hl[1 + c], wtp(W_WD1), nullptr, yv[c], hy[1 + c], kN);

  // 8) a = dense(y2, i0_A), b = dense(y2, i0_B)   (f16 only)
  gemm128<1><<<MT, 256, 0, stream>>>(hy[0], wtp(W_I0A0), fin(P_I0_AB), nullptr, ha[0], kN);
  gemm128<1><<<MT, 256, 0, stream>>>(hy[0], wtp(W_I0B0), fin(P_I0_BB), nullptr, hb[0], kN);
  for (int c = 0; c < 3; ++c) {
    gemm128<1><<<MT, 256, 0, stream>>>(hy[1 + c], wtp(W_I0A1), nullptr, nullptr, ha[1 + c], kN);
    gemm128<1><<<MT, 256, 0, stream>>>(hy[1 + c], wtp(W_I0B1), nullptr, nullptr, hb[1 + c], kN);
  }

  // 9) tensor product -> hl (reused), then out = dense(tp, i0_C) -> t[0..3]
  tp_kernel<<<(int)((NF + 255) / 256), 256, 0, stream>>>(ha[0], ha[1], ha[2], ha[3],
                                                         hb[0], hb[1], hb[2], hb[3],
                                                         hl[0], hl[1], hl[2], hl[3]);
  gemm128<0><<<MT, 256, 0, stream>>>(hl[0], wtp(W_I0C0), fin(P_I0_CB), t[0], nullptr, kN);
  for (int c = 0; c < 3; ++c)
    gemm128<0><<<MT, 256, 0, stream>>>(hl[1 + c], wtp(W_I0C1), nullptr, t[1 + c], nullptr, kN);

  // 10) y3 = y2 + LN2(out)  (in-place f32 scalar, f16 copies in hy)
  ln2_add_kernel<<<kN, kF, 0, stream>>>(t[0], t[1], t[2], t[3], fin(P_N2_G0), fin(P_N2_B0),
                                        fin(P_N2_G1), ys, yv[0], yv[1], yv[2],
                                        hy[0], hy[1], hy[2], hy[3]);

  // 11) i1 interaction: a = dense(y3, i1_A), b = dense(y3, i1_B)
  gemm128<1><<<MT, 256, 0, stream>>>(hy[0], wtp(W_I1A0), fin(P_I1_AB), nullptr, ha[0], kN);
  gemm128<1><<<MT, 256, 0, stream>>>(hy[0], wtp(W_I1B0), fin(P_I1_BB), nullptr, hb[0], kN);
  for (int c = 0; c < 3; ++c) {
    gemm128<1><<<MT, 256, 0, stream>>>(hy[1 + c], wtp(W_I1A1), nullptr, nullptr, ha[1 + c], kN);
    gemm128<1><<<MT, 256, 0, stream>>>(hy[1 + c], wtp(W_I1B1), nullptr, nullptr, hb[1 + c], kN);
  }

  // 12) ts = scalar tensor product; out_s = ts @ i1_C0 + i1_Cb
  ts_kernel<<<(int)((NF + 255) / 256), 256, 0, stream>>>(ha[0], ha[1], ha[2], ha[3],
                                                         hb[0], hb[1], hb[2], hb[3], hl[0]);
  gemm128<0><<<MT, 256, 0, stream>>>(hl[0], wtp(W_I1C0), fin(P_I1_CB), t[0], nullptr, kN);

  // 13) final: d_out = y3_scalar + LN3(out_s)
  ln3_final_kernel<<<kN, kF, 0, stream>>>(t[0], fin(P_N3_G0), fin(P_N3_B0), ys, (float*)d_out);
}
